// TNGen_14963666059366
// MI455X (gfx1250) — matrix operand order repaired and checked by reading, not compiled, not run on hardware
//
#include <hip/hip_runtime.h>
#include <math.h>

#define N_NODES 50000
#define N_EDGES 800000
#define S_SEEDS 8192
#define FOUT 320   // FEAT * NUM_PRED
#define OUTC 321   // 1 (missing) + 320 (feat)

typedef __attribute__((ext_vector_type(16))) __bf16 v16bf;
typedef __attribute__((ext_vector_type(8)))  __bf16 v8bf;
typedef __attribute__((ext_vector_type(4)))  __bf16 v4bf;
typedef __attribute__((ext_vector_type(8)))  float  v8f;

__device__ __forceinline__ int lane_id() { return threadIdx.x & 31; }

// ---------------- WMMA 16x16x32 bf16 fragment helpers (wave32) ----------------
// A fragment (16x32): lane = row m (l&15); half h = l>>4 selects K [16h, 16h+16)
// relative to column offset kk -> two contiguous 8xbf16 runs -> 2x ds_load_b128.
__device__ __forceinline__ v16bf load_a_lds(const __bf16* a, int lda, int kk) {
  const int l = lane_id(), m = l & 15, h = l >> 4;
  const __bf16* p = a + m * lda + kk + 16 * h;
  const v8bf lo = *(const v8bf*)p;
  const v8bf hi = *(const v8bf*)(p + 8);
  v16bf f;
#pragma unroll
  for (int j = 0; j < 8; ++j) { f[j] = lo[j]; f[j + 8] = hi[j]; }
  return f;
}
// B fragment (32x16) from PREPPED column-major bf16 weights Bp[n*K + k]:
// lane = col n (l&15); half h = l>>4 selects K [kk+16h, kk+16h+16) -> 2x global_load_b128.
__device__ __forceinline__ v16bf load_b_kc(const __bf16* bp, int K, int col0, int kk) {
  const int l = lane_id(), n = l & 15, h = l >> 4;
  const __bf16* p = bp + (size_t)(col0 + n) * K + kk + 16 * h;
  const v8bf lo = *(const v8bf*)p;
  const v8bf hi = *(const v8bf*)(p + 8);
  v16bf f;
#pragma unroll
  for (int j = 0; j < 8; ++j) { f[j] = lo[j]; f[j + 8] = hi[j]; }
  return f;
}
__device__ __forceinline__ v8f wmma_bf16(v16bf a, v16bf b, v8f c) {
  return __builtin_amdgcn_wmma_f32_16x16x32_bf16(false, a, false, b, (short)0, c,
                                                 false, false);
}
// C/D layout: VGPR i, lanes 0-15 -> (M=i, N=lane); lanes 16-31 -> (M=8+i, N=lane-16)
__device__ __forceinline__ void store_c_f32(float* d, int ldd, v8f c) {
  const int l = lane_id(), n = l & 15, mo = (l >> 4) * 8;
#pragma unroll
  for (int i = 0; i < 8; ++i) d[(mo + i) * ldd + n] = c[i];
}
__device__ __forceinline__ void store_c_act_bf16(__bf16* d, int ldd, v8f c,
                                                 const float* bias, bool leaky) {
  const int l = lane_id(), n = l & 15, mo = (l >> 4) * 8;
  const float bv = bias[n];
#pragma unroll
  for (int i = 0; i < 8; ++i) {
    float v = c[i] + bv;
    v = (v > 0.f) ? v : (leaky ? 0.01f * v : 0.f);
    d[(mo + i) * ldd + n] = (__bf16)v;
  }
}

// ---------------- Weight prep: f32 row-major (K x N) -> bf16 col-major [n][k] ----------------
__global__ void prep_wT(const float* __restrict__ W, __bf16* __restrict__ O, int K, int N) {
  const int i = blockIdx.x * blockDim.x + threadIdx.x;
  if (i >= K * N) return;
  const int n = i / K, k = i - n * K;
  O[i] = (__bf16)W[(size_t)k * N + n];
}

// ---------------- Graph-conv kernels ----------------
// C(N x 64) = A(N x 64) @ W(64 x 64), W prepped bf16 col-major; 1 wave / 16-row tile.
__global__ __launch_bounds__(32) void gemm_rows16_64x64(
    const float* __restrict__ A, const __bf16* __restrict__ Wp, float* __restrict__ C) {
  __shared__ __align__(16) __bf16 sA[16 * 64];
  const int m0 = blockIdx.x * 16;
  const int tid = threadIdx.x;
  // stage 16x64 fp32 tile -> bf16 LDS (float4 loads, 8-byte LDS stores)
  for (int i = tid; i < 256; i += 32) {
    const float4 v = ((const float4*)(A + (size_t)m0 * 64))[i];
    v4bf o;
    o[0] = (__bf16)v.x; o[1] = (__bf16)v.y; o[2] = (__bf16)v.z; o[3] = (__bf16)v.w;
    ((v4bf*)sA)[i] = o;
  }
  __syncthreads();
  const v16bf a0 = load_a_lds(sA, 64, 0);
  const v16bf a1 = load_a_lds(sA, 64, 32);
#pragma unroll
  for (int t = 0; t < 4; ++t) {
    v8f acc = {};
    acc = wmma_bf16(a0, load_b_kc(Wp, 64, 16 * t, 0), acc);
    acc = wmma_bf16(a1, load_b_kc(Wp, 64, 16 * t, 32), acc);
    store_c_f32(C + (size_t)m0 * 64 + 16 * t, 64, acc);
  }
}

__global__ void zero_kernel(float* __restrict__ p, int n) {
  const int i = blockIdx.x * blockDim.x + threadIdx.x;
  if (i < n) p[i] = 0.f;
}

// One thread per (edge, 16B feature chunk): coalesced float4 gather + 4 float atomics
__global__ void scatter_kernel(const float* __restrict__ P, const int* __restrict__ src,
                               const int* __restrict__ dst, float* __restrict__ agg,
                               float* __restrict__ deg) {
  const int i = blockIdx.x * blockDim.x + threadIdx.x;
  if (i >= N_EDGES * 16) return;
  const int e = i >> 4, c = i & 15;
  const int s = src[e], d = dst[e];
  const float4 v = ((const float4*)(P + (size_t)s * 64))[c];
  float* base = agg + (size_t)d * 64 + 4 * c;
  atomicAdd(base + 0, v.x);
  atomicAdd(base + 1, v.y);
  atomicAdd(base + 2, v.z);
  atomicAdd(base + 3, v.w);
  if (c == 0) atomicAdd(deg + d, 1.0f);
}

__global__ void accum_kernel(float* __restrict__ acc, const float* __restrict__ agg,
                             const float* __restrict__ deg) {
  const int i = blockIdx.x * blockDim.x + threadIdx.x;
  if (i >= N_NODES * 64) return;
  acc[i] += agg[i] / fmaxf(deg[i >> 6], 1.0f);
}

__global__ void relu_bias_kernel(float* __restrict__ h, const float* __restrict__ b) {
  const int i = blockIdx.x * blockDim.x + threadIdx.x;
  if (i >= N_NODES * 64) return;
  h[i] = fmaxf(h[i] + b[i & 63], 0.f);
}

__global__ void leaky_bias_noise_kernel(const float* __restrict__ P,
                                        const float* __restrict__ b,
                                        const float* __restrict__ noise,
                                        float* __restrict__ hout) {
  const int i = blockIdx.x * blockDim.x + threadIdx.x;
  if (i >= N_NODES * 64) return;
  float v = P[i] + b[i & 63];
  v = (v > 0.f) ? v : 0.01f * v;
  hout[i] = v + noise[i];
}

// ---------------- Fused seed-MLP heads ----------------
// grid = (S/16, R); block = 128 (4 waves). z2 (16x2048) streamed through LDS in
// 256-col bf16 chunks; A fragments hoisted into registers and reused.
__global__ __launch_bounds__(128) void head_kernel(
    const float* __restrict__ h,
    const __bf16* __restrict__ pdW1, const float* __restrict__ db1,
    const __bf16* __restrict__ pdW2, const float* __restrict__ db2,
    const float* __restrict__ dW3,   const float* __restrict__ db3,
    const __bf16* __restrict__ pfW1, const float* __restrict__ fb1,
    const __bf16* __restrict__ pfW2, const float* __restrict__ fb2,
    const __bf16* __restrict__ pfW3, const float* __restrict__ fb3,
    float* __restrict__ out) {
  const int r = blockIdx.y;
  const int s0 = blockIdx.x * 16;
  const int wave = threadIdx.x >> 5;
  const int tid = threadIdx.x;

  __shared__ __align__(16) __bf16 sHS[16 * 64];
  __shared__ __align__(16) __bf16 sZ1[16 * 256];
  __shared__ __align__(16) __bf16 sZ2[16 * 256];
  __shared__ __align__(16) __bf16 sZD2[16 * 32];

  // stage seed tile (float4 -> 4xbf16)
  for (int i = tid; i < 256; i += 128) {
    const float4 v = ((const float4*)(h + (size_t)s0 * 64))[i];
    v4bf o;
    o[0] = (__bf16)v.x; o[1] = (__bf16)v.y; o[2] = (__bf16)v.z; o[3] = (__bf16)v.w;
    ((v4bf*)sHS)[i] = o;
  }
  __syncthreads();

  const __bf16* pdW1r = pdW1 + (size_t)r * 256 * 64;    // [256 cols][64 k]
  const __bf16* pdW2r = pdW2 + (size_t)r * 32 * 256;    // [32 cols][256 k]
  const __bf16* pfW1r = pfW1 + (size_t)r * 256 * 64;    // [256 cols][64 k]
  const __bf16* pfW2r = pfW2 + (size_t)r * 2048 * 256;  // [2048 cols][256 k]
  const __bf16* pfW3r = pfW3 + (size_t)r * FOUT * 2048; // [320 cols][2048 k]
  const float* dW3r = dW3 + r * 32;  const float* db3r = db3 + r;
  const float* db1r = db1 + r * 256; const float* db2r = db2 + r * 32;
  const float* fb1r = fb1 + r * 256; const float* fb2r = fb2 + r * 2048;
  const float* fb3r = fb3 + r * FOUT;
  float* outr = out + (size_t)r * S_SEEDS * OUTC;

  const v16bf aHS0 = load_a_lds(sHS, 64, 0);
  const v16bf aHS1 = load_a_lds(sHS, 64, 32);

  // ---- missing head layer 1: leaky(hs @ dW1 + db1) -> sZ1 ----
  for (int t = wave; t < 16; t += 4) {
    v8f acc = {};
    acc = wmma_bf16(aHS0, load_b_kc(pdW1r, 64, 16 * t, 0), acc);
    acc = wmma_bf16(aHS1, load_b_kc(pdW1r, 64, 16 * t, 32), acc);
    store_c_act_bf16(sZ1 + 16 * t, 256, acc, db1r + 16 * t, true);
  }
  __syncthreads();
  if (wave < 2) {  // layer 2: leaky(z @ dW2 + db2), 2 tiles on full-EXEC waves 0,1
    v8f acc = {};
#pragma unroll
    for (int kk = 0; kk < 256; kk += 32)
      acc = wmma_bf16(load_a_lds(sZ1, 256, kk), load_b_kc(pdW2r, 256, 16 * wave, kk), acc);
    store_c_act_bf16(sZD2 + 16 * wave, 32, acc, db2r + 16 * wave, true);
  }
  __syncthreads();
  if (tid < 16) {  // layer 3: relu(z @ dW3 + db3) -> out col 0 (scalar dot)
    float s = db3r[0];
#pragma unroll
    for (int k = 0; k < 32; ++k) s += (float)sZD2[tid * 32 + k] * dW3r[k];
    outr[(s0 + tid) * OUTC + 0] = fmaxf(s, 0.f);
  }

  // ---- feat head layer 1: relu(hs @ fW1 + fb1) -> sZ1 (readers synced above) ----
  for (int t = wave; t < 16; t += 4) {
    v8f acc = {};
    acc = wmma_bf16(aHS0, load_b_kc(pfW1r, 64, 16 * t, 0), acc);
    acc = wmma_bf16(aHS1, load_b_kc(pfW1r, 64, 16 * t, 32), acc);
    store_c_act_bf16(sZ1 + 16 * t, 256, acc, fb1r + 16 * t, false);
  }
  __syncthreads();

  // hoist z1 A-fragments: reused for all 8 chunks x 4 col-tiles
  v16bf aZ1[8];
#pragma unroll
  for (int q = 0; q < 8; ++q) aZ1[q] = load_a_lds(sZ1, 256, 32 * q);

  // ---- fused fW2 -> fW3: stream z2 chunks, accumulate 16x320 out tile ----
  v8f oacc[5] = {};  // wave w owns output col-tiles {w, w+4, w+8, w+12, w+16}
  for (int c0 = 0; c0 < 2048; c0 += 256) {
    if (c0 + 256 < 2048) {  // prefetch next chunk's fW3 k-rows (global_prefetch_b8)
      const __bf16* pf = pfW3r + (size_t)(16 * wave) * 2048 + (c0 + 256);
      __builtin_prefetch(pf + (size_t)lane_id() * 64, 0, 1);
    }
    for (int t = wave; t < 16; t += 4) {
      v8f acc = {};
      const int col = c0 + 16 * t;
#pragma unroll
      for (int kk = 0; kk < 256; kk += 32)
        acc = wmma_bf16(aZ1[kk >> 5], load_b_kc(pfW2r, 256, col, kk), acc);
      store_c_act_bf16(sZ2 + 16 * t, 256, acc, fb2r + col, false);
    }
    __syncthreads();
    v16bf aZ2[8];
#pragma unroll
    for (int q = 0; q < 8; ++q) aZ2[q] = load_a_lds(sZ2, 256, 32 * q);
#pragma unroll
    for (int i = 0; i < 5; ++i) {
      const int ot = wave + 4 * i;
#pragma unroll
      for (int kk = 0; kk < 256; kk += 32)
        oacc[i] = wmma_bf16(aZ2[kk >> 5], load_b_kc(pfW3r, 2048, 16 * ot, c0 + kk), oacc[i]);
    }
    __syncthreads();
  }
  {  // epilogue: tanh(acc + fb3) -> out cols 1..320
    const int l = lane_id(), n = l & 15, mo = (l >> 4) * 8;
#pragma unroll
    for (int i = 0; i < 5; ++i) {
      const int ot = wave + 4 * i;
#pragma unroll
      for (int e = 0; e < 8; ++e)
        outr[(s0 + mo + e) * OUTC + 1 + 16 * ot + n] = tanhf(oacc[i][e] + fb3r[16 * ot + n]);
    }
  }
}

// ---------------- Launcher ----------------
extern "C" void kernel_launch(void* const* d_in, const int* in_sizes, int n_in,
                              void* d_out, int out_size, void* d_ws, size_t ws_size,
                              hipStream_t stream) {
  const float* x     = (const float*)d_in[0];
  const float* noise = (const float*)d_in[1];
  const int* srcs[3] = {(const int*)d_in[3], (const int*)d_in[5], (const int*)d_in[7]};
  const int* dsts[3] = {(const int*)d_in[4], (const int*)d_in[6], (const int*)d_in[8]};
  const float* W1 = (const float*)d_in[9];   const float* b1 = (const float*)d_in[10];
  const float* W2 = (const float*)d_in[11];  const float* b2 = (const float*)d_in[12];
  const float* Wl = (const float*)d_in[13];  const float* bl = (const float*)d_in[14];
  const float* dW1 = (const float*)d_in[15]; const float* db1 = (const float*)d_in[16];
  const float* dW2 = (const float*)d_in[17]; const float* db2 = (const float*)d_in[18];
  const float* dW3 = (const float*)d_in[19]; const float* db3 = (const float*)d_in[20];
  const float* fW1 = (const float*)d_in[21]; const float* fb1 = (const float*)d_in[22];
  const float* fW2 = (const float*)d_in[23]; const float* fb2 = (const float*)d_in[24];
  const float* fW3 = (const float*)d_in[25]; const float* fb3 = (const float*)d_in[26];
  float* out = (float*)d_out;

  float* ws = (float*)d_ws;
  const int NH = N_NODES * 64;
  float* P   = ws;                    // transformed features (N x 64)
  float* agg = ws + 1 * (size_t)NH;   // per-relation aggregate
  float* hA  = ws + 2 * (size_t)NH;   // layer accumulator / hidden state
  float* hB  = ws + 3 * (size_t)NH;
  float* deg = ws + 4 * (size_t)NH;   // N floats

  // bf16 weight pool (16B-aligned: 51.4MB float region is a multiple of 16)
  __bf16* wb = (__bf16*)(ws + 4 * (size_t)NH + N_NODES);
  __bf16* pWc1 = wb;                        // 3 x [64][64]     = 12288
  __bf16* pWc2 = pWc1 + 3 * 64 * 64;        // 12288
  __bf16* pWl  = pWc2 + 3 * 64 * 64;        // [64][64]         = 4096
  __bf16* pdW1 = pWl  + 64 * 64;            // 3 x [256][64]    = 49152
  __bf16* pdW2 = pdW1 + 3 * 256 * 64;       // 3 x [32][256]    = 24576
  __bf16* pfW1 = pdW2 + 3 * 32 * 256;       // 3 x [256][64]    = 49152
  __bf16* pfW2 = pfW1 + 3 * 256 * 64;       // 3 x [2048][256]  = 1572864
  __bf16* pfW3 = pfW2 + 3 * (size_t)2048 * 256;  // 3 x [320][2048] = 1966080

  const int EB = (N_EDGES * 16 + 255) / 256;
  const int NB = (NH + 255) / 256;
  const int DB = (N_NODES + 255) / 256;
  const int GB = N_NODES / 16;  // 50000 % 16 == 0

  // ---- weight prep (f32 row-major (K,N) -> bf16 col-major [n][k]) ----
  for (int r = 0; r < 3; ++r) {
    prep_wT<<<(64 * 64 + 255) / 256, 256, 0, stream>>>(W1 + r * 64 * 64, pWc1 + r * 64 * 64, 64, 64);
    prep_wT<<<(64 * 64 + 255) / 256, 256, 0, stream>>>(W2 + r * 64 * 64, pWc2 + r * 64 * 64, 64, 64);
    prep_wT<<<(64 * 256 + 255) / 256, 256, 0, stream>>>(dW1 + r * 64 * 256, pdW1 + r * 256 * 64, 64, 256);
    prep_wT<<<(256 * 32 + 255) / 256, 256, 0, stream>>>(dW2 + r * 256 * 32, pdW2 + r * 32 * 256, 256, 32);
    prep_wT<<<(64 * 256 + 255) / 256, 256, 0, stream>>>(fW1 + r * 64 * 256, pfW1 + r * 256 * 64, 64, 256);
    prep_wT<<<(256 * 2048 + 255) / 256, 256, 0, stream>>>(fW2 + (size_t)r * 256 * 2048,
                                                          pfW2 + (size_t)r * 2048 * 256, 256, 2048);
    prep_wT<<<(2048 * FOUT + 255) / 256, 256, 0, stream>>>(fW3 + (size_t)r * 2048 * FOUT,
                                                           pfW3 + (size_t)r * FOUT * 2048, 2048, FOUT);
  }
  prep_wT<<<(64 * 64 + 255) / 256, 256, 0, stream>>>(Wl, pWl, 64, 64);

  // ---- conv layer 1 (input x) -> hA ----
  zero_kernel<<<NB, 256, 0, stream>>>(hA, NH);
  for (int r = 0; r < 3; ++r) {
    gemm_rows16_64x64<<<GB, 32, 0, stream>>>(x, pWc1 + r * 64 * 64, P);
    zero_kernel<<<NB, 256, 0, stream>>>(agg, NH);
    zero_kernel<<<DB, 256, 0, stream>>>(deg, N_NODES);
    scatter_kernel<<<EB, 256, 0, stream>>>(P, srcs[r], dsts[r], agg, deg);
    accum_kernel<<<NB, 256, 0, stream>>>(hA, agg, deg);
  }
  relu_bias_kernel<<<NB, 256, 0, stream>>>(hA, b1);

  // ---- conv layer 2 (input hA) -> hB ----
  zero_kernel<<<NB, 256, 0, stream>>>(hB, NH);
  for (int r = 0; r < 3; ++r) {
    gemm_rows16_64x64<<<GB, 32, 0, stream>>>(hA, pWc2 + r * 64 * 64, P);
    zero_kernel<<<NB, 256, 0, stream>>>(agg, NH);
    zero_kernel<<<DB, 256, 0, stream>>>(deg, N_NODES);
    scatter_kernel<<<EB, 256, 0, stream>>>(P, srcs[r], dsts[r], agg, deg);
    accum_kernel<<<NB, 256, 0, stream>>>(hB, agg, deg);
  }
  relu_bias_kernel<<<NB, 256, 0, stream>>>(hB, b2);

  // ---- linear + leaky + noise -> hA (reused) ----
  gemm_rows16_64x64<<<GB, 32, 0, stream>>>(hB, pWl, P);
  leaky_bias_noise_kernel<<<NB, 256, 0, stream>>>(P, bl, noise, hA);

  // ---- fused per-relation seed MLP heads (hs = first 8192 rows of hA) ----
  head_kernel<<<dim3(S_SEEDS / 16, 3), 128, 0, stream>>>(
      hA, pdW1, db1, pdW2, db2, dW3, db3, pfW1, fb1, pfW2, fb2, pfW3, fb3, out);
}
